// GridDownsample_14748917694821
// MI455X (gfx1250) — matrix operand order repaired and checked by reading, not compiled, not run on hardware
//
#include <hip/hip_runtime.h>
#include <stdint.h>

#define GDIM   64
#define NSEG   (GDIM * GDIM * GDIM)   // 262144 segments
#define CIN    96
#define COUT   192
#define NTILES (COUT / 16)            // 12
#define KTILES (CIN / 32)             // 3
#define RT     2                      // row-tiles (of 16 rows) per wave

typedef __attribute__((ext_vector_type(16))) __bf16 v16bf;
typedef __attribute__((ext_vector_type(8)))  __bf16 v8bf;
typedef __attribute__((ext_vector_type(8)))  float  v8f;
typedef __attribute__((ext_vector_type(4)))  float  v4f;

// ---------------- zero counts / coord-sums ----------------
__global__ void gd_zero_kernel(unsigned* counts, float* csum) {
  int i = blockIdx.x * blockDim.x + threadIdx.x;
  if (i < NSEG)     counts[i] = 0u;
  if (i < NSEG * 4) csum[i]   = 0.0f;
}

// ---------------- pack W (fp32 [96,192]) into bf16 WMMA-B fragments ----------------
// B 32x16 bf16 layout: lanes 0-15 elems 0..7 <-> K 0..7, elems 8..15 <-> K 16..23; lanes 16-31
// elems 0..7 <-> K 8..15, elems 8..15 <-> K 24..31; column = lane & 15.  wpack index = (((kt*NTILES+nt)*32)+lane)*16 + e
__global__ void gd_wpack_kernel(const float* __restrict__ W, __bf16* __restrict__ wpack) {
  int t = blockIdx.x * blockDim.x + threadIdx.x;
  if (t >= KTILES * NTILES * 32 * 16) return;
  int e    = t & 15;
  int lane = (t >> 4) & 31;
  int rest = t >> 9;                  // kt*NTILES + nt
  int nt   = rest % NTILES;
  int kt   = rest / NTILES;
  int k    = kt * 32 + ((lane < 16) ? 0 : 8) + (e & 7) + ((e & 8) ? 16 : 0);
  int col  = nt * 16 + (lane & 15);
  wpack[t] = (__bf16)W[k * COUT + col];
}

// ---------------- per-point voxel scatter: counts + coord sums + seg cache ----------------
__global__ void gd_scatter_kernel(const float* __restrict__ cfeat,
                                  const int* __restrict__ coords,
                                  int* __restrict__ seg,
                                  unsigned* __restrict__ counts,
                                  float* __restrict__ csum, int n) {
  int i = blockIdx.x * blockDim.x + threadIdx.x;
  if (i >= n) return;
  int c0 = coords[3 * i + 0] >> 1;
  int c1 = coords[3 * i + 1] >> 1;
  int c2 = coords[3 * i + 2] >> 1;
  int s  = (c0 * GDIM + c1) * GDIM + c2;
  seg[i] = s;
  atomicAdd(&counts[s], 1u);
  atomicAdd(&csum[4 * s + 0], cfeat[4 * i + 0]);
  atomicAdd(&csum[4 * s + 1], cfeat[4 * i + 1]);
  atomicAdd(&csum[4 * s + 2], cfeat[4 * i + 2]);
  atomicAdd(&csum[4 * s + 3], cfeat[4 * i + 3]);
}

// ---------------- seed pooled: -inf for occupied voxels, 0 for empties ----------------
__global__ void gd_prep_pooled_kernel(const unsigned* __restrict__ counts,
                                      float* __restrict__ pooled) {
  long i = (long)blockIdx.x * blockDim.x + threadIdx.x;
  if (i >= (long)NSEG * COUT) return;
  int s = (int)(i / COUT);
  pooled[i] = (counts[s] != 0u) ? -__builtin_inff() : 0.0f;
}

// ---------------- fused LN + bf16 WMMA GEMM + segment-max scatter ----------------
// 1 wave per RT (=2) 16-row tiles: B fragments loaded once per nt serve both tiles.
__global__ void __launch_bounds__(32)
gd_gemm_kernel(const float* __restrict__ feats,
               const float* __restrict__ gamma,
               const float* __restrict__ beta,
               const float* __restrict__ bias,
               const __bf16* __restrict__ wpack,
               const int* __restrict__ seg,
               float* __restrict__ pooled,
               float* __restrict__ dump, int n) {
  __shared__ __align__(16) __bf16 lx[RT * 16 * CIN];  // normalized rows, bf16, stride 96
  __shared__ int lseg[RT * 16];

  const int lane    = threadIdx.x;      // 0..31
  const int m       = lane & 15;        // row within 16-row tile
  const int half    = lane >> 4;        // channel half: 0 -> ch 0..47, 1 -> ch 48..95
  const int rowbase = blockIdx.x * (RT * 16);

  lseg[lane] = (rowbase + lane < n) ? seg[rowbase + lane] : -1;

  // ---- LayerNorm: lane pair (m, m+16) splits a row's 96 channels 48/48 ----
#pragma unroll
  for (int t = 0; t < RT; ++t) {
    const int row = rowbase + t * 16 + m;
    float v[48];
    float s = 0.f, q = 0.f;
    if (row < n) {
      const float* src = feats + (size_t)row * CIN + half * 48;
#pragma unroll
      for (int i = 0; i < 12; ++i) {
        v4f tv = *(const v4f*)(src + 4 * i);
#pragma unroll
        for (int j = 0; j < 4; ++j) { float x = tv[j]; v[4 * i + j] = x; s += x; q += x * x; }
      }
    } else {
#pragma unroll
      for (int i = 0; i < 48; ++i) v[i] = 0.f;
    }
    float s_all = s + __shfl_xor(s, 16, 32);
    float q_all = q + __shfl_xor(q, 16, 32);
    float mean  = s_all * (1.0f / CIN);
    float var   = q_all * (1.0f / CIN) - mean * mean;
    float rstd  = rsqrtf(var + 1e-5f);
#pragma unroll
    for (int i = 0; i < 48; ++i) {
      int ch = half * 48 + i;
      float xn = (v[i] - mean) * rstd * gamma[ch] + beta[ch];
      lx[(t * 16 + m) * CIN + ch] = (__bf16)xn;
    }
  }
  __syncthreads();

  // ---- A fragments: 16-bit A 16x32 layout ----
  // lanes 0-15:  elems 0..7 <-> K k0+0..7,  elems 8..15 <-> K k0+16..23
  // lanes 16-31: elems 0..7 <-> K k0+8..15, elems 8..15 <-> K k0+24..31
  v16bf afrag[RT][KTILES];
#pragma unroll
  for (int t = 0; t < RT; ++t) {
#pragma unroll
    for (int kt = 0; kt < KTILES; ++kt) {
      const __bf16* bp = &lx[(t * 16 + m) * CIN + kt * 32 + half * 8];
      v8bf lo = *(const v8bf*)bp;         // 16B ds_read
      v8bf hi = *(const v8bf*)(bp + 16);  // 16B ds_read
      v16bf a;
#pragma unroll
      for (int j = 0; j < 8; ++j) { a[j] = lo[j]; a[8 + j] = hi[j]; }
      afrag[t][kt] = a;
    }
  }

  // ---- segment base pointers, read ONCE into registers (branchless via dump slot) ----
  // C/D layout: VGPR g, lanes 0-15 -> row g, lanes 16-31 -> row 8+g; col = lane&15
  float* baseptr[RT][8];
#pragma unroll
  for (int t = 0; t < RT; ++t) {
#pragma unroll
    for (int g = 0; g < 8; ++g) {
      int sg = lseg[t * 16 + g + half * 8];
      baseptr[t][g] = (sg >= 0) ? (pooled + (size_t)sg * COUT + m) : (dump + m);
    }
  }

  // ---- 12 N-tiles: load 3 B fragments once, run RT x 3 WMMAs, scatter maxes ----
#pragma unroll
  for (int nt = 0; nt < NTILES; ++nt) {
    v16bf bfrag[KTILES];
#pragma unroll
    for (int kt = 0; kt < KTILES; ++kt)
      bfrag[kt] = *(const v16bf*)(wpack + (((size_t)kt * NTILES + nt) * 32 + lane) * 16);
    float bc = bias[nt * 16 + m];
#pragma unroll
    for (int t = 0; t < RT; ++t) {
      v8f acc = {};  // SRC2 can be inline 0 per ISA; bias added post-WMMA
#pragma unroll
      for (int kt = 0; kt < KTILES; ++kt)
        acc = __builtin_amdgcn_wmma_f32_16x16x32_bf16(
            false, afrag[t][kt], false, bfrag[kt], (short)0, acc, false, false);
#pragma unroll
      for (int g = 0; g < 8; ++g) {
        // nt*16 is a compile-time constant per unrolled iter -> folds into offset: field
        (void)__hip_atomic_fetch_max(baseptr[t][g] + nt * 16, acc[g] + bc,
                                     __ATOMIC_RELAXED, __HIP_MEMORY_SCOPE_AGENT);
      }
    }
  }
}

// ---------------- coords_down = csum / max(count,1) ----------------
__global__ void gd_cdown_kernel(const unsigned* __restrict__ counts,
                                const float* __restrict__ csum,
                                float* __restrict__ cdown) {
  int i = blockIdx.x * blockDim.x + threadIdx.x;
  if (i >= NSEG * 4) return;
  int s = i >> 2;
  cdown[i] = csum[i] / fmaxf((float)counts[s], 1.0f);
}

extern "C" void kernel_launch(void* const* d_in, const int* in_sizes, int n_in,
                              void* d_out, int out_size, void* d_ws, size_t ws_size,
                              hipStream_t stream) {
  const float* feats  = (const float*)d_in[0];  // [N, 96]
  const float* cfeat  = (const float*)d_in[1];  // [N, 4]
  const int*   coords = (const int*)d_in[2];    // [N, 3]
  const float* gamma  = (const float*)d_in[3];  // [96]
  const float* beta   = (const float*)d_in[4];  // [96]
  const float* W      = (const float*)d_in[5];  // [96, 192]
  const float* bias   = (const float*)d_in[6];  // [192]
  const int n = in_sizes[0] / CIN;              // 1,000,000

  float* pooled = (float*)d_out;                        // [NSEG, 192]
  float* cdown  = pooled + (size_t)NSEG * COUT;         // [NSEG, 4]

  // workspace carve-up (256B aligned slabs)
  auto align_up = [](size_t x) { return (x + 255) & ~(size_t)255; };
  char* ws = (char*)d_ws;
  int* seg = (int*)ws;                    ws += align_up((size_t)n * sizeof(int));
  unsigned* counts = (unsigned*)ws;       ws += align_up((size_t)NSEG * sizeof(unsigned));
  float* csum = (float*)ws;               ws += align_up((size_t)NSEG * 4 * sizeof(float));
  __bf16* wpack = (__bf16*)ws;            ws += align_up((size_t)KTILES * NTILES * 32 * 16 * sizeof(__bf16));
  float* dump = (float*)ws;               // 192-float sink for invalid-row atomics

  const int tpb = 256;
  const long pooled_total = (long)NSEG * COUT;

  gd_zero_kernel<<<(NSEG * 4 + tpb - 1) / tpb, tpb, 0, stream>>>(counts, csum);

  gd_wpack_kernel<<<(KTILES * NTILES * 32 * 16 + tpb - 1) / tpb, tpb, 0, stream>>>(
      W, wpack);

  gd_scatter_kernel<<<(n + tpb - 1) / tpb, tpb, 0, stream>>>(
      cfeat, coords, seg, counts, csum, n);

  gd_prep_pooled_kernel<<<(unsigned)((pooled_total + tpb - 1) / tpb), tpb, 0, stream>>>(
      counts, pooled);

  gd_gemm_kernel<<<(n + RT * 16 - 1) / (RT * 16), 32, 0, stream>>>(
      feats, gamma, beta, bias, wpack, seg, pooled, dump, n);

  gd_cdown_kernel<<<(NSEG * 4 + tpb - 1) / tpb, tpb, 0, stream>>>(
      counts, csum, cdown);
}
